// SiameseChangeDetectorSNN_9594956939286
// MI455X (gfx1250) — matrix operand order repaired and checked by reading, not compiled, not run on hardware
//
#include <hip/hip_runtime.h>

// ---------------------------------------------------------------------------
// SiameseChangeDetectorSNN for MI455X (gfx1250, wave32, WMMA).
//
// conv2 feeds only global-average-pool + BN + threshold (all linear before the
// threshold), so mean(BN(conv2(s)))[n] = (Sum_{c,tap} w2[n,c,tap]*S_c(tap)) /
// 16384 * scale2[n] + bias2[n], where S_c(tap) = spike-map channel sum over
// the valid shifted window = T_c - excludedRow - excludedCol + excludedCorner.
// Spikes are exactly {0,1}: S_c reduces to 9 integer counts per channel,
// accumulated with integer atomics (exact and deterministic).
//
// Kernel 1: conv1 as implicit GEMM (v_wmma_f32_16x16x32_f16, K = 90 -> 96),
//   fused BN -> 2x2 maxpool -> spike -> 9-way spike statistics. The GEMM K
//   axis is permuted as k' = tap*10 + c so every A-fragment VGPR pair is a
//   single aligned ds_load_b32 from the [row][col][chan] staging tile.
// Kernel 2: tiny per-batch finish (S reconstruction, 128x576 matvec, BN,
//   spike, |s0-s1|, fc1+relu, fc2).
// ---------------------------------------------------------------------------

typedef __attribute__((ext_vector_type(16))) _Float16     v16h;
typedef __attribute__((ext_vector_type(8)))  _Float16     v8h;
typedef __attribute__((ext_vector_type(8)))  float        v8f;
typedef __attribute__((ext_vector_type(8)))  unsigned int v8u;

#define BN_EPS 1e-5f

// LDS staging geometry for kernel 1
#define XS_ROWS 6      // conv rows y0-1 .. y0+4
#define XS_COLS 258    // cols -1 .. 256
#define XS_CD   12     // channels 0..9 real, 10..11 zeroed (K padding target)
#define W_KS    104    // padded K stride of transposed weight matrix [64][96]

// Offset (in halves) of GEMM-K element k' inside the staged tile, relative to
// laneBase = (rr*XS_COLS + colbase)*XS_CD.   k' = tap*10 + c, tap = ky*3+kx.
// k' >= 90 (padding) maps to the zeroed channel slots 10/11 of a safe cell.
__host__ __device__ constexpr int AOFF(int k) {
  return (k < 90) ? (((k / 10) / 3) * (XS_COLS * XS_CD)   // ky -> row
                   + ((k / 10) % 3) * XS_CD               // kx -> col
                   + (k % 10))                            // c
                  : (10 + (k & 1));
}

// counts layout: (img 2, b 32, ch 64, 9):  0=T 1=row0 2=row127 3=col0 4=col127
// 5=corner(0,0) 6=corner(0,127) 7=corner(127,0) 8=corner(127,127)
#define NSUMS (2 * 32 * 64 * 9)

__global__ void snn_zero_kernel(int* __restrict__ p, int n) {
  int i = blockIdx.x * blockDim.x + threadIdx.x;
  if (i < n) p[i] = 0;
}

// grid: (64 rowTiles, 32 batch, 2 img), block: 512 threads (16 waves).
// Each block: conv rows 4*tileY .. 4*tileY+3 (pooled rows 2*tileY, 2*tileY+1)
// over the full 256-column width; wave w owns conv cols 16w..16w+15.
__global__ __launch_bounds__(512) void snn_conv1_kernel(
    const float* __restrict__ x0, const float* __restrict__ x1,
    const float* __restrict__ w1,
    const float* __restrict__ g1, const float* __restrict__ b1,
    const float* __restrict__ m1, const float* __restrict__ v1,
    int* __restrict__ sums) {
  __shared__ __align__(16) _Float16 xs[XS_ROWS * XS_COLS * XS_CD];
  __shared__ __align__(16) _Float16 wmat[64 * W_KS];
  __shared__ float bnS[64];
  __shared__ float bnB[64];
  __shared__ int   ldsT[64];

  const int tid   = threadIdx.x;
  const int tileY = blockIdx.x;        // 0..63
  const int b     = blockIdx.y;        // 0..31
  const int img   = blockIdx.z;        // 0..1
  const float* __restrict__ x = (img == 0) ? x0 : x1;
  const int y0base = 4 * tileY;

  // ---- stage input tile as f16, column-fastest for coalesced HBM reads ----
  // (channels 10,11 and out-of-range pixels -> 0)
  for (int i = tid; i < XS_CD * XS_ROWS * XS_COLS; i += 512) {
    const int s = i % XS_COLS;
    const int r = (i / XS_COLS) % XS_ROWS;
    const int c = i / (XS_COLS * XS_ROWS);
    const int y  = y0base - 1 + r;
    const int xc = s - 1;
    float val = 0.0f;
    if (c < 10 && y >= 0 && y < 256 && xc >= 0 && xc < 256)
      val = x[(((b * 10 + c) * 256) + y) * 256 + xc];
    xs[(r * XS_COLS + s) * XS_CD + c] = (_Float16)val;
  }
  // ---- stage transposed weights with permuted K: wmat[n][tap*10+c] ----
  for (int i = tid; i < 64 * 96; i += 512) {
    const int k = i % 96;
    const int n = i / 96;
    float val = (k < 90) ? w1[n * 90 + (k % 10) * 9 + (k / 10)] : 0.0f;
    wmat[n * W_KS + k] = (_Float16)val;
  }
  if (tid < 64) {
    const float s = g1[tid] * rsqrtf(v1[tid] + BN_EPS);
    bnS[tid] = s;
    bnB[tid] = b1[tid] - m1[tid] * s;
    ldsT[tid] = 0;
  }
  __syncthreads();

  const int lane = tid & 31;
  const int wave = tid >> 5;           // 0..15 -> conv col chunk
  const int mlo  = lane & 15;
  const int hi   = lane >> 4;          // lane half
  const int colbase = wave * 16 + mlo; // staged col of this lane's A row (M=mlo)
  int* __restrict__ gsum = sums + (img * 32 + b) * 64 * 9;

  // Per-lane A-fragment offsets (halves), hoisted out of all loops.
  // VGPR pair v of K-chunk kt holds k' = kt*32 + (v<4 ? 2v : 16+2(v-4)) (+8 hi).
  int aoffs[3][8];
#pragma unroll
  for (int kt = 0; kt < 3; ++kt)
#pragma unroll
    for (int vg = 0; vg < 8; ++vg) {
      const int kA = kt * 32 + (vg < 4 ? 2 * vg : 16 + 2 * (vg - 4));
      aoffs[kt][vg] = hi ? AOFF(kA + 8) : AOFF(kA);
    }

  const v8f vzero = {0.f, 0.f, 0.f, 0.f, 0.f, 0.f, 0.f, 0.f};

#pragma unroll
  for (int rp = 0; rp < 2; ++rp) {     // pooled row within the block
    v8f acc[2][4];
#pragma unroll
    for (int dy = 0; dy < 2; ++dy)
#pragma unroll
      for (int ct = 0; ct < 4; ++ct) acc[dy][ct] = vzero;

    int laneBaseA[2];
#pragma unroll
    for (int dy = 0; dy < 2; ++dy)
      laneBaseA[dy] = ((2 * rp + dy) * XS_COLS + colbase) * XS_CD;

#pragma unroll
    for (int kt = 0; kt < 3; ++kt) {   // K = 96 in chunks of 32
      // A fragments: 8 aligned b32 LDS loads each (channel pairs c,c+1)
      v16h afr[2];
#pragma unroll
      for (int dy = 0; dy < 2; ++dy) {
        v8u au;
#pragma unroll
        for (int vg = 0; vg < 8; ++vg)
          au[vg] = *(const unsigned int*)&xs[laneBaseA[dy] + aoffs[kt][vg]];
        afr[dy] = __builtin_bit_cast(v16h, au);
      }
#pragma unroll
      for (int ct = 0; ct < 4; ++ct) { // 64 out channels as 4 N-tiles
        // B fragment: lane = N, lane half h holds K 8h..8h+7 and 16+8h..16+8h+7
        const _Float16* wp = &wmat[(ct * 16 + mlo) * W_KS + kt * 32 + hi * 8];
        const v8h blo = *(const v8h*)(wp);
        const v8h bhi = *(const v8h*)(wp + 16);
        const v16h bfr = __builtin_shufflevector(
            blo, bhi, 0, 1, 2, 3, 4, 5, 6, 7, 8, 9, 10, 11, 12, 13, 14, 15);
        acc[0][ct] = __builtin_amdgcn_wmma_f32_16x16x32_f16(
            false, afr[0], false, bfr, (short)0, acc[0][ct], false, false);
        acc[1][ct] = __builtin_amdgcn_wmma_f32_16x16x32_f16(
            false, afr[1], false, bfr, (short)0, acc[1][ct], false, false);
      }
    }

    // ---- epilogue: BN -> 2x2 maxpool -> spike -> spike statistics ----
    // D layout: N = mlo, M = vgpr + 8*hi  ->  conv x = wave*16 + 8*hi + vgpr
    const int py  = tileY * 2 + rp;     // pooled row 0..127
    const int px0 = wave * 8 + hi * 4;  // pooled col of element p=0
#pragma unroll
    for (int ct = 0; ct < 4; ++ct) {
      const int ch = ct * 16 + mlo;
      const float sc = bnS[ch], bi = bnB[ch];
      int sp[4];
      int localT = 0;
#pragma unroll
      for (int p = 0; p < 4; ++p) {
        const float v00 = fmaf(acc[0][ct][2 * p],     sc, bi);
        const float v01 = fmaf(acc[0][ct][2 * p + 1], sc, bi);
        const float v10 = fmaf(acc[1][ct][2 * p],     sc, bi);
        const float v11 = fmaf(acc[1][ct][2 * p + 1], sc, bi);
        const float mx  = fmaxf(fmaxf(v00, v01), fmaxf(v10, v11));
        sp[p] = (mx > 1.0f) ? 1 : 0;
        localT += sp[p];
      }
      atomicAdd(&ldsT[ch], localT);                        // total T (per block)
      if (py == 0)   atomicAdd(&gsum[ch * 9 + 1], localT); // row 0
      if (py == 127) atomicAdd(&gsum[ch * 9 + 2], localT); // row 127
      if (px0 == 0) {                                      // px == 0 at p=0
        atomicAdd(&gsum[ch * 9 + 3], sp[0]);               // col 0
        if (py == 0)   atomicAdd(&gsum[ch * 9 + 5], sp[0]); // (0,0)
        if (py == 127) atomicAdd(&gsum[ch * 9 + 7], sp[0]); // (127,0)
      }
      if (px0 == 124) {                                    // px == 127 at p=3
        atomicAdd(&gsum[ch * 9 + 4], sp[3]);               // col 127
        if (py == 0)   atomicAdd(&gsum[ch * 9 + 6], sp[3]); // (0,127)
        if (py == 127) atomicAdd(&gsum[ch * 9 + 8], sp[3]); // (127,127)
      }
    }
  }

  __syncthreads();
  if (tid < 64) atomicAdd(&gsum[tid * 9 + 0], ldsT[tid]);
}

// grid: 32 (batch), block: 128 threads (thread n = conv2 out channel).
__global__ __launch_bounds__(128) void snn_finish_kernel(
    const int* __restrict__ sums, const float* __restrict__ w2,
    const float* __restrict__ g2, const float* __restrict__ b2,
    const float* __restrict__ m2, const float* __restrict__ v2,
    const float* __restrict__ fc1w, const float* __restrict__ fc1b,
    const float* __restrict__ fc2w, const float* __restrict__ fc2b,
    float* __restrict__ out) {
  __shared__ float feat[128];
  __shared__ float hbuf[64];
  const int b = blockIdx.x;
  const int n = threadIdx.x;

  const int* __restrict__ gA = sums + ((0 * 32 + b) * 64) * 9;
  const int* __restrict__ gB = sums + ((1 * 32 + b) * 64) * 9;
  const float* __restrict__ wrow = w2 + n * 576;

  float z0 = 0.0f, z1 = 0.0f;
  for (int c = 0; c < 64; ++c) {
    float S0[9], S1[9];
#pragma unroll
    for (int im = 0; im < 2; ++im) {
      const int* g = (im == 0) ? (gA + c * 9) : (gB + c * 9);
      const float T  = (float)g[0];
      const float r0 = (float)g[1], rR = (float)g[2];
      const float c0 = (float)g[3], cR = (float)g[4];
      const float k00 = (float)g[5], k0R = (float)g[6];
      const float kR0 = (float)g[7], kRR = (float)g[8];
      float* S = (im == 0) ? S0 : S1;
      S[0] = T - rR - cR + kRR;  // ky=0,kx=0: rows 0..126, cols 0..126
      S[1] = T - rR;             // ky=0,kx=1
      S[2] = T - rR - c0 + kR0;  // ky=0,kx=2: rows 0..126, cols 1..127
      S[3] = T - cR;             // ky=1,kx=0
      S[4] = T;                  // ky=1,kx=1
      S[5] = T - c0;             // ky=1,kx=2
      S[6] = T - r0 - cR + k0R;  // ky=2,kx=0: rows 1..127, cols 0..126
      S[7] = T - r0;             // ky=2,kx=1
      S[8] = T - r0 - c0 + k00;  // ky=2,kx=2
    }
#pragma unroll
    for (int t = 0; t < 9; ++t) {
      const float w = wrow[c * 9 + t];
      z0 = fmaf(w, S0[t], z0);
      z1 = fmaf(w, S1[t], z1);
    }
  }

  const float sc = g2[n] * rsqrtf(v2[n] + BN_EPS);
  const float bi = b2[n] - m2[n] * sc;
  const float mA = fmaf(z0 * (1.0f / 16384.0f), sc, bi);
  const float mB = fmaf(z1 * (1.0f / 16384.0f), sc, bi);
  const float sA = (mA > 1.0f) ? 1.0f : 0.0f;
  const float sB = (mB > 1.0f) ? 1.0f : 0.0f;
  feat[n] = fabsf(sA - sB);
  __syncthreads();

  if (n < 64) {
    float a = fc1b[n];
    for (int j = 0; j < 128; ++j) a = fmaf(fc1w[n * 128 + j], feat[j], a);
    hbuf[n] = fmaxf(a, 0.0f);
  }
  __syncthreads();

  if (n < 5) {
    float a = fc2b[n];
    for (int j = 0; j < 64; ++j) a = fmaf(fc2w[n * 64 + j], hbuf[j], a);
    out[b * 5 + n] = a;
  }
}

extern "C" void kernel_launch(void* const* d_in, const int* in_sizes, int n_in,
                              void* d_out, int out_size, void* d_ws, size_t ws_size,
                              hipStream_t stream) {
  const float* x0   = (const float*)d_in[0];
  const float* x1   = (const float*)d_in[1];
  const float* c1w  = (const float*)d_in[2];
  const float* g1   = (const float*)d_in[3];
  const float* b1   = (const float*)d_in[4];
  const float* m1   = (const float*)d_in[5];
  const float* v1   = (const float*)d_in[6];
  const float* c2w  = (const float*)d_in[7];
  const float* g2   = (const float*)d_in[8];
  const float* b2   = (const float*)d_in[9];
  const float* m2   = (const float*)d_in[10];
  const float* v2   = (const float*)d_in[11];
  const float* fc1w = (const float*)d_in[12];
  const float* fc1b = (const float*)d_in[13];
  const float* fc2w = (const float*)d_in[14];
  const float* fc2b = (const float*)d_in[15];
  float* out = (float*)d_out;
  int* sums = (int*)d_ws;  // needs 2*32*64*9*4 = 147456 bytes of workspace

  snn_zero_kernel<<<(NSUMS + 255) / 256, 256, 0, stream>>>(sums, NSUMS);
  snn_conv1_kernel<<<dim3(64, 32, 2), 512, 0, stream>>>(
      x0, x1, c1w, g1, b1, m1, v1, sums);
  snn_finish_kernel<<<32, 128, 0, stream>>>(
      sums, c2w, g2, b2, m2, v2, fc1w, fc1b, fc2w, fc2b, out);
}
